// ParallelAttention_11811160064680
// MI455X (gfx1250) — matrix operand order repaired and checked by reading, not compiled, not run on hardware
//
#include <hip/hip_runtime.h>
#include <hip/hip_bf16.h>

typedef __bf16 bf16_t;
typedef __attribute__((ext_vector_type(16))) __bf16 v16bf;
typedef __attribute__((ext_vector_type(8)))  float  v8f;

union F16x16 { v16bf v; uint4 u[2]; };

constexpr int B_ = 2, S_ = 2048, E_ = 2048, H_ = 32, KVH_ = 8, D_ = 64;
constexpr int HD_  = H_ * D_;     // 2048
constexpr int KVD_ = KVH_ * D_;   // 512

__device__ __forceinline__ v8f wmma_bf16(const F16x16& a, const F16x16& b, v8f c) {
  // D = A(16x32 bf16) * B(32x16 bf16) + C(16x16 f32)
  return __builtin_amdgcn_wmma_f32_16x16x32_bf16(false, a.v, false, b.v,
                                                 (short)0, c, false, false);
}

// CDNA5 async copy: global -> LDS, 16 bytes per lane, tracked by ASYNCcnt.
// GV addressing: VDST = LDS byte address (low 32 bits of generic pointer),
// VADDR = 64-bit global address pair, SADDR = off.
__device__ __forceinline__ void async_g2l_b128(const bf16_t* g, bf16_t* l) {
  unsigned int lds_addr = (unsigned int)(unsigned long long)(const void*)l;
  unsigned long long gaddr = (unsigned long long)(const void*)g;
  asm volatile("global_load_async_to_lds_b128 %0, %1, off"
               :: "v"(lds_addr), "v"(gaddr) : "memory");
}
__device__ __forceinline__ void wait_async0() {
  asm volatile("s_wait_asynccnt 0" ::: "memory");
}

// ---------------------------------------------------------------- converts
__global__ void k_f32_to_bf16(const float* __restrict__ in,
                              bf16_t* __restrict__ out, int n) {
  int i = blockIdx.x * blockDim.x + threadIdx.x;
  if (i < n) out[i] = (bf16_t)in[i];
}

// ---------------------------------------------------------------- GEMM
// C[M,N] = A[M,K] (bf16 row-major) @ W[K,N] (bf16 row-major)
// Block: 256 threads (8 waves, 4x2), tile 128x128, K-step 32.
template <typename OutT>
__global__ __launch_bounds__(256) void k_gemm_bf16(
    const bf16_t* __restrict__ A, const bf16_t* __restrict__ W,
    OutT* __restrict__ C, int M, int N, int K) {
  __shared__ bf16_t As[128][40];   // [m][k], +8 pad keeps 16B-aligned rows
  __shared__ bf16_t Bt[128][40];   // [n][k] (transposed during staging)

  const int tid    = threadIdx.x;
  const int lane   = tid & 31;
  const int wave   = tid >> 5;
  const int lane16 = lane & 15;
  const int hlf    = lane >> 4;
  const int wm = wave >> 1, wn = wave & 1;

  const int blockM = blockIdx.y * 128;
  const int blockN = blockIdx.x * 128;
  const int waveM  = wm * 32;
  const int waveN  = wn * 64;

  v8f acc[2][4];
  const v8f vzero = {0.f, 0.f, 0.f, 0.f, 0.f, 0.f, 0.f, 0.f};
#pragma unroll
  for (int i = 0; i < 2; ++i)
#pragma unroll
    for (int j = 0; j < 4; ++j) acc[i][j] = vzero;

  const int aRow = tid >> 1, aCol = (tid & 1) * 16;
  const int bK = tid & 31, bN0 = (tid >> 5) * 16;

  for (int k0 = 0; k0 < K; k0 += 32) {
    // ---- stage A tile 128x32 asynchronously (global -> LDS, no VGPR bounce)
    const bf16_t* ap = A + (size_t)(blockM + aRow) * K + k0 + aCol;
    async_g2l_b128(ap,     &As[aRow][aCol]);
    async_g2l_b128(ap + 8, &As[aRow][aCol + 8]);
    // ---- stage W tile 32x128, transposed into Bt[n][k] (VGPR bounce needed)
    const bf16_t* wp = W + (size_t)(k0 + bK) * N + blockN + bN0;
    __attribute__((aligned(16))) bf16_t tmp[16];
    *(uint4*)tmp       = *(const uint4*)wp;
    *(uint4*)(tmp + 8) = *(const uint4*)(wp + 8);
#pragma unroll
    for (int e = 0; e < 16; ++e) Bt[bN0 + e][bK] = tmp[e];
    if (k0 + 32 < K) {   // hint next K-slab into cache (global_prefetch)
      __builtin_prefetch(ap + 32, 0, 1);
      __builtin_prefetch(wp + (size_t)32 * N, 0, 1);
    }
    wait_async0();
    __syncthreads();

    // ---- A fragments: lanes 0-15 hold K 0-15; lanes 16-31 hold 16-31
    F16x16 af[2];
#pragma unroll
    for (int mt = 0; mt < 2; ++mt) {
      const bf16_t* p = &As[waveM + mt * 16 + lane16][hlf * 16];
      af[mt].u[0] = *(const uint4*)p;
      af[mt].u[1] = *(const uint4*)(p + 8);
    }
    // ---- B fragments: N=lane16, K = hlf*16 + e (contiguous in Bt row)
#pragma unroll
    for (int nt = 0; nt < 4; ++nt) {
      F16x16 bfr;
      const bf16_t* p = &Bt[waveN + nt * 16 + lane16][hlf * 16];
      bfr.u[0] = *(const uint4*)p;
      bfr.u[1] = *(const uint4*)(p + 8);
#pragma unroll
      for (int mt = 0; mt < 2; ++mt)
        acc[mt][nt] = wmma_bf16(af[mt], bfr, acc[mt][nt]);
    }
    __syncthreads();
  }

  // ---- store: C-layout VGPR v -> M = v + hlf*8, N = lane16
#pragma unroll
  for (int mt = 0; mt < 2; ++mt)
#pragma unroll
    for (int nt = 0; nt < 4; ++nt)
#pragma unroll
      for (int v = 0; v < 8; ++v) {
        int r = blockM + waveM + mt * 16 + v + hlf * 8;
        int c = blockN + waveN + nt * 16 + lane16;
        C[(size_t)r * N + c] = (OutT)acc[mt][nt][v];
      }
}

// ---------------------------------------------------------------- RoPE
__global__ void k_rope(bf16_t* __restrict__ buf, int NH, int n) {
  int i = blockIdx.x * blockDim.x + threadIdx.x;
  if (i >= n) return;
  int d   = i & 31;                 // rotary pair index 0..31
  int h   = (i >> 5) % NH;
  int row = i / (32 * NH);
  int s   = row & (S_ - 1);
  float freq = __powf(10000.0f, -(float)d * (1.0f / 32.0f));
  float sn, cs;
  __sincosf((float)s * freq, &sn, &cs);
  bf16_t* p = buf + (size_t)row * (NH * 64) + h * 64 + d;
  float t1 = (float)p[0], t2 = (float)p[32];
  p[0]  = (bf16_t)(t1 * cs - t2 * sn);
  p[32] = (bf16_t)(t1 * sn + t2 * cs);
}

// ---------------------------------------------------------------- attention
// Grid (S/64, H, B); 128 threads = 4 waves, wave w owns 16 query rows.
__global__ __launch_bounds__(128) void k_attn(
    const bf16_t* __restrict__ Q, const bf16_t* __restrict__ Kb,
    const bf16_t* __restrict__ V, bf16_t* __restrict__ O) {
  __shared__ bf16_t ks[64][72];        // K rows  [key][d]
  __shared__ bf16_t vt[64][72];        // V^T     [d][key]
  __shared__ bf16_t pw[4][16][40];     // wave-private P tiles [m][key]

  const int tid    = threadIdx.x;
  const int lane   = tid & 31;
  const int wave   = tid >> 5;
  const int lane16 = lane & 15;
  const int hlf    = lane >> 4;

  const int qblk = blockIdx.x, h = blockIdx.y, b = blockIdx.z;
  const int hk = h >> 2;               // GQA: 4 Q heads per KV head
  const int q0 = qblk * 64 + wave * 16;

  // Q A-fragments resident for the whole KV sweep (2 chunks of K=32 over D=64)
  F16x16 qf[2];
  {
    const bf16_t* qrow = Q + (size_t)(b * S_ + q0 + lane16) * HD_ + h * D_;
#pragma unroll
    for (int c = 0; c < 2; ++c) {
      const bf16_t* p = qrow + c * 32 + hlf * 16;
      qf[c].u[0] = *(const uint4*)p;
      qf[c].u[1] = *(const uint4*)(p + 8);
    }
  }

  float mx[8], ls[8];
#pragma unroll
  for (int v = 0; v < 8; ++v) { mx[v] = -3.0e38f; ls[v] = 0.0f; }
  const v8f vzero = {0.f, 0.f, 0.f, 0.f, 0.f, 0.f, 0.f, 0.f};
  v8f oacc[4];
#pragma unroll
  for (int nt = 0; nt < 4; ++nt) oacc[nt] = vzero;

  const int nblk = qblk + 1;           // causal: only KV blocks with j0 <= q
  for (int jb = 0; jb < nblk; ++jb) {
    const int j0 = jb * 64;
    // ---- staging: K rows via async global->LDS DMA; V transposed via VGPRs
    {
      const int key = tid >> 1;
      const int d0  = (tid & 1) * 32;
      const bf16_t* kp = Kb + (size_t)(b * S_ + j0 + key) * KVD_ + hk * D_ + d0;
      async_g2l_b128(kp,      &ks[key][d0]);
      async_g2l_b128(kp + 8,  &ks[key][d0 + 8]);
      async_g2l_b128(kp + 16, &ks[key][d0 + 16]);
      async_g2l_b128(kp + 24, &ks[key][d0 + 24]);
      const bf16_t* vp = V + (size_t)(b * S_ + j0 + key) * KVD_ + hk * D_ + d0;
      __attribute__((aligned(16))) bf16_t tmp[32];
#pragma unroll
      for (int q8 = 0; q8 < 4; ++q8)
        *(uint4*)(tmp + q8 * 8) = *(const uint4*)(vp + q8 * 8);
#pragma unroll
      for (int e = 0; e < 32; ++e) vt[d0 + e][key] = tmp[e];
    }
    wait_async0();
    __syncthreads();

#pragma unroll
    for (int kk = 0; kk < 64; kk += 32) {
      // ---- scores S = Q @ K^T for 32 keys (two 16-key WMMA column tiles)
      v8f s0 = vzero, s1 = vzero;
#pragma unroll
      for (int c = 0; c < 2; ++c) {
        F16x16 b0, b1;
        const bf16_t* p0 = &ks[kk + lane16][c * 32 + hlf * 16];
        b0.u[0] = *(const uint4*)p0; b0.u[1] = *(const uint4*)(p0 + 8);
        const bf16_t* p1 = &ks[kk + 16 + lane16][c * 32 + hlf * 16];
        b1.u[0] = *(const uint4*)p1; b1.u[1] = *(const uint4*)(p1 + 8);
        s0 = wmma_bf16(qf[c], b0, s0);
        s1 = wmma_bf16(qf[c], b1, s1);
      }
      // ---- online softmax over this 32-key slab
      const float scl = 0.125f;        // 1/sqrt(64)
      const int key0 = j0 + kk + lane16;
      const int key1 = key0 + 16;
      float p0v[8], p1v[8], alpha[8];
#pragma unroll
      for (int v = 0; v < 8; ++v) {
        const int qr = q0 + v + hlf * 8;
        float a0 = (key0 <= qr) ? s0[v] * scl : -3.0e38f;
        float a1 = (key1 <= qr) ? s1[v] * scl : -3.0e38f;
        float rm = fmaxf(a0, a1);
#pragma unroll
        for (int off = 1; off < 16; off <<= 1)
          rm = fmaxf(rm, __shfl_xor(rm, off, 32));   // 16-lane halves
        const float mnew = fmaxf(mx[v], rm);
        alpha[v] = __expf(mx[v] - mnew);
        mx[v] = mnew;
        p0v[v] = __expf(a0 - mnew);
        p1v[v] = __expf(a1 - mnew);
        float rs = p0v[v] + p1v[v];
#pragma unroll
        for (int off = 1; off < 16; off <<= 1)
          rs += __shfl_xor(rs, off, 32);
        ls[v] = ls[v] * alpha[v] + rs;
      }
#pragma unroll
      for (int nt = 0; nt < 4; ++nt)
#pragma unroll
        for (int v = 0; v < 8; ++v) oacc[nt][v] *= alpha[v];

      // ---- relayout P (C-layout -> A-layout) through wave-private LDS
#pragma unroll
      for (int v = 0; v < 8; ++v) {
        pw[wave][v + hlf * 8][lane16]      = (bf16_t)p0v[v];
        pw[wave][v + hlf * 8][16 + lane16] = (bf16_t)p1v[v];
      }
      asm volatile("s_wait_dscnt 0" ::: "memory");   // wave-local RAW on LDS

      F16x16 pa;
      const bf16_t* pp = &pw[wave][lane16][hlf * 16];
      pa.u[0] = *(const uint4*)pp;
      pa.u[1] = *(const uint4*)(pp + 8);
      // ---- O += P @ V  (contraction over 32 keys, 4 d-tiles)
#pragma unroll
      for (int nt = 0; nt < 4; ++nt) {
        F16x16 vf;
        const bf16_t* vp2 = &vt[nt * 16 + lane16][kk + hlf * 16];
        vf.u[0] = *(const uint4*)vp2;
        vf.u[1] = *(const uint4*)(vp2 + 8);
        oacc[nt] = wmma_bf16(pa, vf, oacc[nt]);
      }
    }
    __syncthreads();
  }

  // ---- normalize and emit bf16 attention output [B*S][H*D]
#pragma unroll
  for (int v = 0; v < 8; ++v) ls[v] = 1.0f / ls[v];
#pragma unroll
  for (int nt = 0; nt < 4; ++nt)
#pragma unroll
    for (int v = 0; v < 8; ++v) {
      const int qr = q0 + v + hlf * 8;
      O[(size_t)(b * S_ + qr) * HD_ + h * D_ + nt * 16 + lane16] =
          (bf16_t)(oacc[nt][v] * ls[v]);
    }
}

// ---------------------------------------------------------------- launch
extern "C" void kernel_launch(void* const* d_in, const int* in_sizes, int n_in,
                              void* d_out, int out_size, void* d_ws, size_t ws_size,
                              hipStream_t stream) {
  const float* x  = (const float*)d_in[0];
  const float* Wq = (const float*)d_in[1];
  const float* Wk = (const float*)d_in[2];
  const float* Wv = (const float*)d_in[3];
  const float* Wo = (const float*)d_in[4];
  float* out = (float*)d_out;

  char* ws = (char*)d_ws;                 // ~80 MB used
  bf16_t* xb  = (bf16_t*)(ws);            // B*S*E          16 MB
  bf16_t* wqb = (bf16_t*)(ws + 16777216); // E*2048          8 MB
  bf16_t* wkb = (bf16_t*)(ws + 25165824); // E*512           2 MB
  bf16_t* wvb = (bf16_t*)(ws + 27262976); // E*512           2 MB
  bf16_t* wob = (bf16_t*)(ws + 29360128); // 2048*E          8 MB
  bf16_t* qb  = (bf16_t*)(ws + 37748736); // B*S*2048 (rope) 16 MB
  bf16_t* kb  = (bf16_t*)(ws + 54525952); // B*S*512  (rope)  4 MB
  bf16_t* vb  = (bf16_t*)(ws + 58720256); // B*S*512          4 MB
  bf16_t* ob  = (bf16_t*)(ws + 62914560); // B*S*2048        16 MB

  const int nx  = B_ * S_ * E_;
  const int nwq = E_ * HD_;
  const int nwk = E_ * KVD_;
  const int nwo = HD_ * E_;
  k_f32_to_bf16<<<(nx  + 255) / 256, 256, 0, stream>>>(x,  xb,  nx);
  k_f32_to_bf16<<<(nwq + 255) / 256, 256, 0, stream>>>(Wq, wqb, nwq);
  k_f32_to_bf16<<<(nwk + 255) / 256, 256, 0, stream>>>(Wk, wkb, nwk);
  k_f32_to_bf16<<<(nwk + 255) / 256, 256, 0, stream>>>(Wv, wvb, nwk);
  k_f32_to_bf16<<<(nwo + 255) / 256, 256, 0, stream>>>(Wo, wob, nwo);

  const int M = B_ * S_;  // 4096
  k_gemm_bf16<bf16_t><<<dim3(HD_  / 128, M / 128), 256, 0, stream>>>(xb, wqb, qb, M, HD_,  E_);
  k_gemm_bf16<bf16_t><<<dim3(KVD_ / 128, M / 128), 256, 0, stream>>>(xb, wkb, kb, M, KVD_, E_);
  k_gemm_bf16<bf16_t><<<dim3(KVD_ / 128, M / 128), 256, 0, stream>>>(xb, wvb, vb, M, KVD_, E_);

  const int nq = M * H_ * 32;
  const int nk = M * KVH_ * 32;
  k_rope<<<(nq + 255) / 256, 256, 0, stream>>>(qb, H_,   nq);
  k_rope<<<(nk + 255) / 256, 256, 0, stream>>>(kb, KVH_, nk);

  k_attn<<<dim3(S_ / 64, H_, B_), 128, 0, stream>>>(qb, kb, vb, ob);

  k_gemm_bf16<float><<<dim3(E_ / 128, M / 128), 256, 0, stream>>>(ob, wob, out, M, E_, HD_);
}
